// ParallelExperts_3685081940646
// MI455X (gfx1250) — matrix operand order repaired and checked by reading, not compiled, not run on hardware
//
#include <hip/hip_runtime.h>
#include <hip/hip_bf16.h>

// ---------------------------------------------------------------------------
// MoE grouped GEMM for MI455X (gfx1250, wave32):
//   out[off_e : off_e+size_e] = X[off_e : off_e+size_e] @ W[e]
// fp32 data, computed via bf16x3 split-precision WMMA (near-fp32 accuracy):
//   x = xh + xl (two bf16), x*y ~= xh*yh + xh*yl + xl*yh
// ---------------------------------------------------------------------------

typedef __attribute__((ext_vector_type(16))) __bf16 v16bf;
typedef __attribute__((ext_vector_type(8)))  __bf16 v8bf;
typedef __attribute__((ext_vector_type(8)))  float  v8f;

#define DIN   4096
#define DOUT  4096
#define BM    128
#define BN    128
#define BK    32
#define BSTR  40          // BK + 8 pad: bank-conflict-free, keeps 16B alignment
#define NTHR  256         // 8 waves of 32

static __device__ __forceinline__ unsigned short f32_to_bf16_rne(float f) {
  union { float f; unsigned u; } v; v.f = f;
  unsigned r = v.u + 0x7FFFu + ((v.u >> 16) & 1u);   // round-to-nearest-even
  return (unsigned short)(r >> 16);
}
static __device__ __forceinline__ float bf16_to_f32(unsigned short h) {
  union { unsigned u; float f; } v; v.u = ((unsigned)h) << 16;
  return v.f;
}
static __device__ __forceinline__ void split_bf16(float f, unsigned short& h,
                                                  unsigned short& l) {
  h = f32_to_bf16_rne(f);
  l = f32_to_bf16_rne(f - bf16_to_f32(h));
}

__global__ __launch_bounds__(NTHR)
void moe_gemm_bf16x3(const float* __restrict__ X,
                     const long long* __restrict__ esz,
                     const float* __restrict__ W,
                     float* __restrict__ Out) {
  // LDS: double buffered hi/lo tiles.  A: M-major [BM][BSTR]; B: N-major [BN][BSTR]
  __shared__ __align__(16) unsigned short sAh[2][BM * BSTR];
  __shared__ __align__(16) unsigned short sAl[2][BM * BSTR];
  __shared__ __align__(16) unsigned short sBh[2][BN * BSTR];
  __shared__ __align__(16) unsigned short sBl[2][BN * BSTR];

  const int tid  = threadIdx.x;
  const int e    = blockIdx.z;
  const int n0   = blockIdx.x * BN;
  const int me   = blockIdx.y * BM;            // row tile inside this expert

  // expert row offset / size (uniform per block)
  long long off = 0;
  for (int i = 0; i < e; ++i) off += esz[i];
  const long long sz = esz[e];
  if ((long long)me >= sz) return;             // uniform early-out, no barriers yet

  const float* __restrict__ Wb = W + (size_t)e * DIN * DOUT;

  // wave decomposition: 8 waves -> 2 (M) x 4 (N); wave tile 64x32 -> 4x2 frags
  const int wid  = tid >> 5;
  const int lane = tid & 31;
  const int wm   = wid & 1;                    // 0..1   -> 64-row half
  const int wn   = wid >> 1;                   // 0..3   -> 32-col quarter
  const int lm   = lane & 15;                  // row (A) / col (B) within frag
  const int kh   = lane >> 4;                  // k-half selector

  v8f c[4][2];
#pragma unroll
  for (int i = 0; i < 4; ++i)
#pragma unroll
    for (int j = 0; j < 2; ++j)
#pragma unroll
      for (int r = 0; r < 8; ++r) c[i][j][r] = 0.0f;

  const int S = DIN / BK;                      // 128 k-stages
  float4 ra[4], rb[4];

  // ---- prologue: global loads for stage 0 -------------------------------
#pragma unroll
  for (int i = 0; i < 4; ++i) {
    const int idx = tid + i * NTHR;            // 0..1023
    const int row = idx >> 3, kq = idx & 7;    // A: 128 rows x 8 quads
    ra[i] = (me + row < sz)
              ? *(const float4*)(X + (size_t)(off + me + row) * DIN + kq * 4)
              : make_float4(0.f, 0.f, 0.f, 0.f);
    const int kr = idx >> 5, nq = idx & 31;    // B: 32 k-rows x 32 quads
    rb[i] = *(const float4*)(Wb + (size_t)kr * DOUT + n0 + nq * 4);
  }

  for (int s = 0; s < S; ++s) {
    const int buf = s & 1;

    // ---- stage regs -> LDS (split hi/lo) --------------------------------
#pragma unroll
    for (int i = 0; i < 4; ++i) {
      const int idx = tid + i * NTHR;
      // A tile: vectorized 4x bf16 (8B) stores, M-major
      {
        const int row = idx >> 3, kq = idx & 7;
        unsigned short h[4], l[4];
        split_bf16(ra[i].x, h[0], l[0]); split_bf16(ra[i].y, h[1], l[1]);
        split_bf16(ra[i].z, h[2], l[2]); split_bf16(ra[i].w, h[3], l[3]);
        uint2 ph, pl;
        ph.x = (unsigned)h[0] | ((unsigned)h[1] << 16);
        ph.y = (unsigned)h[2] | ((unsigned)h[3] << 16);
        pl.x = (unsigned)l[0] | ((unsigned)l[1] << 16);
        pl.y = (unsigned)l[2] | ((unsigned)l[3] << 16);
        *(uint2*)&sAh[buf][row * BSTR + kq * 4] = ph;
        *(uint2*)&sAl[buf][row * BSTR + kq * 4] = pl;
      }
      // B tile: transpose into N-major via scattered b16 stores
      {
        const int kr = idx >> 5, nq = idx & 31;
        unsigned short h[4], l[4];
        split_bf16(rb[i].x, h[0], l[0]); split_bf16(rb[i].y, h[1], l[1]);
        split_bf16(rb[i].z, h[2], l[2]); split_bf16(rb[i].w, h[3], l[3]);
#pragma unroll
        for (int j = 0; j < 4; ++j) {
          sBh[buf][(nq * 4 + j) * BSTR + kr] = h[j];
          sBl[buf][(nq * 4 + j) * BSTR + kr] = l[j];
        }
      }
    }
    __syncthreads();

    // ---- issue global loads for stage s+1 (latency hidden by compute) ---
    if (s + 1 < S) {
      const int kt = (s + 1) * BK;
#pragma unroll
      for (int i = 0; i < 4; ++i) {
        const int idx = tid + i * NTHR;
        const int row = idx >> 3, kq = idx & 7;
        ra[i] = (me + row < sz)
                  ? *(const float4*)(X + (size_t)(off + me + row) * DIN + kt + kq * 4)
                  : make_float4(0.f, 0.f, 0.f, 0.f);
        const int kr = idx >> 5, nq = idx & 31;
        const float* wp = Wb + (size_t)(kt + kr) * DOUT + n0 + nq * 4;
        rb[i] = *(const float4*)wp;
        if (kt + BK < DIN)                    // global_prefetch_b8, one stage ahead
          __builtin_prefetch(wp + (size_t)BK * DOUT, 0, 3);
      }
    }

    // ---- WMMA compute over LDS stage ------------------------------------
    // B frags: lane = col n (lm), k-half kh -> 16 contiguous bf16 (N-major)
    v16bf bh[2], bl[2];
#pragma unroll
    for (int j = 0; j < 2; ++j) {
      const int nrow = wn * 32 + j * 16 + lm;
      const unsigned short* ph = &sBh[buf][nrow * BSTR + kh * 16];
      const unsigned short* pl = &sBl[buf][nrow * BSTR + kh * 16];
      v8bf b0 = *(const v8bf*)(ph), b1 = *(const v8bf*)(ph + 8);
      bh[j] = __builtin_shufflevector(b0, b1, 0,1,2,3,4,5,6,7,8,9,10,11,12,13,14,15);
      v8bf c0 = *(const v8bf*)(pl), c1 = *(const v8bf*)(pl + 8);
      bl[j] = __builtin_shufflevector(c0, c1, 0,1,2,3,4,5,6,7,8,9,10,11,12,13,14,15);
    }
#pragma unroll
    for (int i = 0; i < 4; ++i) {
      // A frag: lane = row m (lm); VGPR0-7 = K kh*16..kh*16+15 (contiguous)
      const int arow = wm * 64 + i * 16 + lm;
      const unsigned short* ph = &sAh[buf][arow * BSTR];
      const unsigned short* pl = &sAl[buf][arow * BSTR];
      v8bf a0 = *(const v8bf*)(ph + kh * 16);
      v8bf a1 = *(const v8bf*)(ph + kh * 16 + 8);
      v16bf ah = __builtin_shufflevector(a0, a1, 0,1,2,3,4,5,6,7,8,9,10,11,12,13,14,15);
      v8bf a2 = *(const v8bf*)(pl + kh * 16);
      v8bf a3 = *(const v8bf*)(pl + kh * 16 + 8);
      v16bf al = __builtin_shufflevector(a2, a3, 0,1,2,3,4,5,6,7,8,9,10,11,12,13,14,15);
#pragma unroll
      for (int j = 0; j < 2; ++j) {
        // small terms first, then the dominant hi*hi term
        c[i][j] = __builtin_amdgcn_wmma_f32_16x16x32_bf16(
            false, al, false, bh[j], (short)0, c[i][j], false, false);
        c[i][j] = __builtin_amdgcn_wmma_f32_16x16x32_bf16(
            false, ah, false, bl[j], (short)0, c[i][j], false, false);
        c[i][j] = __builtin_amdgcn_wmma_f32_16x16x32_bf16(
            false, ah, false, bh[j], (short)0, c[i][j], false, false);
      }
    }
    __syncthreads();
  }

  // ---- epilogue: C layout = VGPR r, lanes0-15 M=r, lanes16-31 M=8+r -----
#pragma unroll
  for (int i = 0; i < 4; ++i) {
    const int rbase = me + wm * 64 + i * 16 + kh * 8;
#pragma unroll
    for (int j = 0; j < 2; ++j) {
      const int col = n0 + wn * 32 + j * 16 + lm;
#pragma unroll
      for (int r = 0; r < 8; ++r) {
        const long long gr = rbase + r;
        if (gr < sz) Out[(size_t)(off + gr) * DOUT + col] = c[i][j][r];
      }
    }
  }
}

extern "C" void kernel_launch(void* const* d_in, const int* in_sizes, int n_in,
                              void* d_out, int out_size, void* d_ws, size_t ws_size,
                              hipStream_t stream) {
  const float*     X   = (const float*)d_in[0];
  const long long* esz = (const long long*)d_in[1];
  const float*     W   = (const float*)d_in[2];
  float*           out = (float*)d_out;

  const int E   = in_sizes[1];                       // 8
  const long long T = (long long)in_sizes[0] / DIN;  // 16384
  const int tpe = (int)(T / E);                      // 2048 rows per expert

  dim3 grid(DOUT / BN, (tpe + BM - 1) / BM, E);
  hipLaunchKernelGGL(moe_gemm_bf16x3, grid, dim3(NTHR), 0, stream,
                     X, esz, W, out);
}
